// Block_38173669327037
// MI455X (gfx1250) — matrix operand order repaired and checked by reading, not compiled, not run on hardware
//
#include <hip/hip_runtime.h>

typedef float    v8f  __attribute__((ext_vector_type(8)));
typedef float    v4f  __attribute__((ext_vector_type(4)));
typedef _Float16 v16h __attribute__((ext_vector_type(16)));
typedef _Float16 h8   __attribute__((ext_vector_type(8)));

#define TT   4
#define BB   32
#define CC   384
#define NN   196
#define NPAD 224
#define HHD  12
#define DH   32
#define HIDD 1536
#define EPSB 1e-5f

// ---------- WMMA fragment helpers (16x16x32 f16, wave32 layouts per CDNA5 ISA 7.12.2) ----------

// A-fragment: M=row per lane (lane%16), K chunks {k0+8g..+7, k0+16+8g..+7}, row-major src.
__device__ __forceinline__ v16h frag_rows(const _Float16* __restrict__ p, int row, int ld,
                                          int k0, int g) {
  const h8 lo = *(const h8*)(p + (size_t)row * ld + k0 + 8 * g);
  const h8 hi = *(const h8*)(p + (size_t)row * ld + k0 + 16 + 8 * g);
  return __builtin_shufflevector(lo, hi, 0,1,2,3,4,5,6,7,8,9,10,11,12,13,14,15);
}
// B-fragment: N=col per lane (lane%16), K chunks {k0+8g..+7, k0+16+8g..+7}; src holds B^T row-major.
__device__ __forceinline__ v16h frag_cols(const _Float16* __restrict__ p, int col, int ld,
                                          int k0, int g) {
  const h8 lo = *(const h8*)(p + (size_t)col * ld + k0 + 8 * g);
  const h8 hi = *(const h8*)(p + (size_t)col * ld + k0 + 16 + 8 * g);
  return __builtin_shufflevector(lo, hi, 0,1,2,3,4,5,6,7,8,9,10,11,12,13,14,15);
}
__device__ __forceinline__ v8f wmma16(v16h a, v16h b, v8f c) {
  return __builtin_amdgcn_wmma_f32_16x16x32_f16(false, a, false, b, (short)0, c, false, false);
}
__device__ __forceinline__ v8f zero8() {
  return v8f{0.f, 0.f, 0.f, 0.f, 0.f, 0.f, 0.f, 0.f};
}

// ---------- Fused GEMM + BatchNorm + 4-step LIF ----------
// X: [T][B][NPAD][Cin] f16 (token-major). W: [Cout][Cin] f16.
// OUT_T=true : D = W x X, store spikes to [T][B][NPAD][Cout] (token-major, b128 stores)
// OUT_T=false: D = X x W^T, store spikes to [T][B][Cout][NPAD] (channel-major, b128 stores)
// RESID (OUT_T only): output = spike + resid[n][o] (f16), for the x + o residual.
template <bool OUT_T, bool RESID>
__global__ __launch_bounds__(256) void gemm_bn_lif(
    const _Float16* __restrict__ W, const _Float16* __restrict__ X,
    const float* __restrict__ sc, const float* __restrict__ of,
    const _Float16* __restrict__ resid, _Float16* __restrict__ Y,
    int Cin, int Cout, float thr) {
  const int lane = threadIdx.x & 31;
  const int wv   = threadIdx.x >> 5;
  const int g = lane >> 4, l = lane & 15;
  const int nt = blockIdx.x * 16;
  const int ot = (blockIdx.y * 8 + wv) * 16;
  const int b  = blockIdx.z;

  // hoisted per-timestep activation base pointers
  const _Float16* Xt[TT];
#pragma unroll
  for (int t = 0; t < TT; ++t) Xt[t] = X + (size_t)(t * BB + b) * NPAD * Cin;

  v8f acc[TT];
#pragma unroll
  for (int t = 0; t < TT; ++t) acc[t] = zero8();

  for (int k0 = 0; k0 < Cin; k0 += 32) {
    if (OUT_T) {
      const v16h aw = frag_rows(W, ot + l, Cin, k0, g);
#pragma unroll
      for (int t = 0; t < TT; ++t) {
        const v16h bx = frag_cols(Xt[t], nt + l, Cin, k0, g);
        acc[t] = wmma16(aw, bx, acc[t]);
      }
    } else {
      const v16h bw = frag_cols(W, ot + l, Cin, k0, g);
#pragma unroll
      for (int t = 0; t < TT; ++t) {
        const v16h ax = frag_rows(Xt[t], nt + l, Cin, k0, g);
        acc[t] = wmma16(ax, bw, acc[t]);
      }
    }
  }

  if (OUT_T) {
    // lane holds column n = nt+l, rows o = ot + 8g + r
    const int n = nt + l;
    const v4f sc0 = *(const v4f*)(sc + ot + 8 * g);
    const v4f sc1 = *(const v4f*)(sc + ot + 8 * g + 4);
    const v4f of0 = *(const v4f*)(of + ot + 8 * g);
    const v4f of1 = *(const v4f*)(of + ot + 8 * g + 4);
    _Float16 spk[TT][8];
#pragma unroll
    for (int r = 0; r < 8; ++r) {
      const float scr = (r < 4) ? sc0[r & 3] : sc1[r & 3];
      const float ofr = (r < 4) ? of0[r & 3] : of1[r & 3];
      float v = 0.f;
#pragma unroll
      for (int t = 0; t < TT; ++t) {
        const float y = acc[t][r] * scr + ofr;
        v = 0.5f * (v + y);                       // tau = 2
        const bool s = (v >= thr);
        spk[t][r] = s ? (_Float16)1.0f : (_Float16)0.0f;
        if (s) v = 0.f;                           // hard reset
      }
    }
#pragma unroll
    for (int t = 0; t < TT; ++t) {
      h8 out;
#pragma unroll
      for (int r = 0; r < 8; ++r) {
        _Float16 val = spk[t][r];
        if (RESID)
          val = (_Float16)((float)val +
                (float)resid[((size_t)(t * BB + b) * NPAD + n) * Cout + ot + 8 * g + r]);
        out[r] = (n < NN) ? val : (_Float16)0.0f; // zero the n-padding
      }
      *(h8*)(Y + ((size_t)(t * BB + b) * NPAD + n) * Cout + ot + 8 * g) = out;
    }
  } else {
    // lane holds column o = ot+l, rows n = nt + 8g + r
    const int o = ot + l;
    const float scr = sc[o], ofr = of[o];
    h8 outs[TT];
#pragma unroll
    for (int r = 0; r < 8; ++r) {
      const int n = nt + 8 * g + r;
      float v = 0.f;
#pragma unroll
      for (int t = 0; t < TT; ++t) {
        const float y = acc[t][r] * scr + ofr;
        v = 0.5f * (v + y);
        const bool s = (v >= thr);
        outs[t][r] = (s && n < NN) ? (_Float16)1.0f : (_Float16)0.0f;
        if (s) v = 0.f;
      }
    }
#pragma unroll
    for (int t = 0; t < TT; ++t)
      *(h8*)(Y + ((size_t)(t * BB + b) * Cout + o) * NPAD + nt + 8 * g) = outs[t];
  }
}

// ---------- Fused attention: attnT = K*Q^T -> policy mask -> f16 -> *V -> *0.25 -> LIF(0.5) ----------
// qs,ks: [T][B][NPAD][C] f16 spikes; vs: [T][B][C][NPAD] f16 spikes; out oas: [T][B][NPAD][C].
__global__ __launch_bounds__(128) void attn_kernel(
    const _Float16* __restrict__ qs, const _Float16* __restrict__ ks,
    const _Float16* __restrict__ vs, const float* __restrict__ policy,
    _Float16* __restrict__ oas) {
  __shared__ alignas(16) float s_pr[TT][NPAD];
  const int b = blockIdx.z;
  for (int i = threadIdx.x; i < TT * NPAD; i += 128) {
    const int t = i / NPAD, m = i % NPAD;
    s_pr[t][m] = (m < NN) ? policy[(size_t)(t * BB + b) * NN + m] : 0.f;
  }
  __syncthreads();

  const int lane = threadIdx.x & 31, wv = threadIdx.x >> 5;
  const int g = lane >> 4, l = lane & 15;
  const int h  = blockIdx.y * 4 + wv;
  const int nt = blockIdx.x * 16;
  const int n_col = nt + l;

  v8f oacc[TT][2];
#pragma unroll
  for (int t = 0; t < TT; ++t) { oacc[t][0] = zero8(); oacc[t][1] = zero8(); }

#pragma unroll
  for (int t = 0; t < TT; ++t) {
    const size_t tb = (size_t)(t * BB + b);
    const _Float16* q_tb  = qs + tb * NPAD * CC;
    const _Float16* k_tb  = ks + tb * NPAD * CC;
    const _Float16* v_tbh = vs + (tb * CC + (size_t)h * DH) * NPAD;
    const v16h qb = frag_cols(q_tb, n_col, CC, h * DH, g);   // B: cols n, K = d (32 = one WMMA)
    for (int mm = 0; mm < NPAD; mm += 32) {
      const v16h ka0 = frag_rows(k_tb, mm + l,      CC, h * DH, g);
      const v16h ka1 = frag_rows(k_tb, mm + 16 + l, CC, h * DH, g);
      const v8f at0 = wmma16(ka0, qb, zero8());              // attnT tile, m rows mm..+15
      const v8f at1 = wmma16(ka1, qb, zero8());              // attnT tile, m rows mm+16..+31
      // Unconditional vector LDS reads of the policy (4 x ds_load_b128),
      // then apply the diagonal via selects -- no exec-mask churn.
      const float* pr_t = &s_pr[t][0];
      const v4f p0a = *(const v4f*)(pr_t + mm + 8 * g);
      const v4f p0b = *(const v4f*)(pr_t + mm + 8 * g + 4);
      const v4f p1a = *(const v4f*)(pr_t + mm + 16 + 8 * g);
      const v4f p1b = *(const v4f*)(pr_t + mm + 16 + 8 * g + 4);
      // D-fragment of attnT == A-fragment for the second WMMA: mask, convert in-lane.
      v16h a2;
#pragma unroll
      for (int r = 0; r < 8; ++r) {
        const int m0 = mm + 8 * g + r, m1 = mm + 16 + 8 * g + r;
        float f0 = (r < 4) ? p0a[r & 3] : p0b[r & 3];
        float f1 = (r < 4) ? p1a[r & 3] : p1b[r & 3];
        f0 = (m0 == n_col) ? 1.f : f0;                       // pr + (1-pr)*eye
        f1 = (m1 == n_col) ? 1.f : f1;
        a2[r]     = (_Float16)(at0[r] * f0);
        a2[8 + r] = (_Float16)(at1[r] * f1);
      }
#pragma unroll
      for (int dt = 0; dt < 2; ++dt) {
        const v16h vb = frag_cols(v_tbh, dt * 16 + l, NPAD, mm, g);  // cols d, K = m
        oacc[t][dt] = wmma16(a2, vb, oacc[t][dt]);
      }
    }
  }
  // epilogue: *SCALE then LIF(thr=0.5); store spikes token-major [n][c]
#pragma unroll
  for (int dt = 0; dt < 2; ++dt) {
    const int c = h * DH + dt * 16 + l;
#pragma unroll
    for (int r = 0; r < 8; ++r) {
      const int n = nt + 8 * g + r;
      float v = 0.f;
#pragma unroll
      for (int t = 0; t < TT; ++t) {
        const float y = oacc[t][dt][r] * 0.25f;
        v = 0.5f * (v + y);
        const bool s = (v >= 0.5f);
        oas[((size_t)(t * BB + b) * NPAD + n) * CC + c] =
            (s && n < NN) ? (_Float16)1.0f : (_Float16)0.0f;
        if (s) v = 0.f;
      }
    }
  }
}

// ---------- small prep / finish kernels ----------
__global__ void cvt_x(const float* __restrict__ x, _Float16* __restrict__ xh, int total) {
  const int i = blockIdx.x * blockDim.x + threadIdx.x;
  if (i >= total) return;
  const int c = i % CC, n = (i / CC) % NPAD, tb = i / (CC * NPAD);
  xh[i] = (n < NN) ? (_Float16)x[((size_t)tb * CC + c) * NN + n] : (_Float16)0.f;
}
__global__ void cvt_w(const float* __restrict__ w, _Float16* __restrict__ wh, int total) {
  const int i = blockIdx.x * blockDim.x + threadIdx.x;
  if (i < total) wh[i] = (_Float16)w[i];
}
__global__ void bn_prep(const float* __restrict__ gmm, const float* __restrict__ bet,
                        const float* __restrict__ mn, const float* __restrict__ vr,
                        const float* __restrict__ bias, float* __restrict__ sc,
                        float* __restrict__ of, int cout) {
  const int i = blockIdx.x * blockDim.x + threadIdx.x;
  if (i >= cout) return;
  const float inv = gmm[i] * rsqrtf(vr[i] + EPSB);
  sc[i] = inv;
  const float bi = bias ? bias[i] : 0.f;
  of[i] = bet[i] + (bi - mn[i]) * inv;   // conv bias folded through BN
}
__global__ void final_add(const _Float16* __restrict__ x1h, const _Float16* __restrict__ h2s,
                          float* __restrict__ out, int total) {
  const int i = blockIdx.x * blockDim.x + threadIdx.x;
  if (i >= total) return;
  const int n = i % NN, c = (i / NN) % CC, tb = i / (NN * CC);
  const size_t j = ((size_t)tb * NPAD + n) * CC + c;
  out[i] = (float)x1h[j] + (float)h2s[j];
}

// ---------- driver ----------
extern "C" void kernel_launch(void* const* d_in, const int* in_sizes, int n_in,
                              void* d_out, int out_size, void* d_ws, size_t ws_size,
                              hipStream_t stream) {
  (void)in_sizes; (void)n_in; (void)out_size; (void)ws_size;
  const float* x      = (const float*)d_in[0];
  const float* policy = (const float*)d_in[1];
  const float* wq = (const float*)d_in[2];
  const float* qg = (const float*)d_in[3];  const float* qb = (const float*)d_in[4];
  const float* qm = (const float*)d_in[5];  const float* qv = (const float*)d_in[6];
  const float* wk = (const float*)d_in[7];
  const float* kg = (const float*)d_in[8];  const float* kb = (const float*)d_in[9];
  const float* km = (const float*)d_in[10]; const float* kvv = (const float*)d_in[11];
  const float* wv = (const float*)d_in[12];
  const float* vg = (const float*)d_in[13]; const float* vb = (const float*)d_in[14];
  const float* vm = (const float*)d_in[15]; const float* vvv = (const float*)d_in[16];
  const float* wp = (const float*)d_in[17]; const float* bp = (const float*)d_in[18];
  const float* pg = (const float*)d_in[19]; const float* pb = (const float*)d_in[20];
  const float* pm = (const float*)d_in[21]; const float* pv = (const float*)d_in[22];
  const float* w1 = (const float*)d_in[23]; const float* b1 = (const float*)d_in[24];
  const float* g1 = (const float*)d_in[25]; const float* be1 = (const float*)d_in[26];
  const float* m1 = (const float*)d_in[27]; const float* v1 = (const float*)d_in[28];
  const float* w2 = (const float*)d_in[29]; const float* b2 = (const float*)d_in[30];
  const float* g2 = (const float*)d_in[31]; const float* be2 = (const float*)d_in[32];
  const float* m2 = (const float*)d_in[33]; const float* v2 = (const float*)d_in[34];

  char* ws = (char*)d_ws;
  size_t off = 0;
  auto alloc = [&](size_t bytes) -> void* {
    off = (off + 255) & ~(size_t)255;
    void* p = ws + off;
    off += bytes;
    return p;
  };
  const size_t S_act = (size_t)TT * BB * NPAD * CC;     // halves
  const size_t S_hid = (size_t)TT * BB * NPAD * HIDD;

  _Float16* xh  = (_Float16*)alloc(S_act * 2);
  _Float16* qsb = (_Float16*)alloc(S_act * 2);
  _Float16* ksb = (_Float16*)alloc(S_act * 2);
  _Float16* vsb = (_Float16*)alloc(S_act * 2);   // channel-major
  _Float16* oas = (_Float16*)alloc(S_act * 2);
  _Float16* x1h = (_Float16*)alloc(S_act * 2);
  _Float16* h2s = (_Float16*)alloc(S_act * 2);
  _Float16* h1s = (_Float16*)alloc(S_hid * 2);
  _Float16* wqh = (_Float16*)alloc((size_t)CC * CC * 2);
  _Float16* wkh = (_Float16*)alloc((size_t)CC * CC * 2);
  _Float16* wvh = (_Float16*)alloc((size_t)CC * CC * 2);
  _Float16* wph = (_Float16*)alloc((size_t)CC * CC * 2);
  _Float16* w1h = (_Float16*)alloc((size_t)HIDD * CC * 2);
  _Float16* w2h = (_Float16*)alloc((size_t)CC * HIDD * 2);
  float* qsc = (float*)alloc(CC * 4);   float* qof = (float*)alloc(CC * 4);
  float* ksc = (float*)alloc(CC * 4);   float* kof = (float*)alloc(CC * 4);
  float* vsc = (float*)alloc(CC * 4);   float* vof = (float*)alloc(CC * 4);
  float* psc = (float*)alloc(CC * 4);   float* pof = (float*)alloc(CC * 4);
  float* s1  = (float*)alloc(HIDD * 4); float* o1  = (float*)alloc(HIDD * 4);
  float* s2  = (float*)alloc(CC * 4);   float* o2  = (float*)alloc(CC * 4);

  // prep
  cvt_x<<<(int)((S_act + 255) / 256), 256, 0, stream>>>(x, xh, (int)S_act);
  cvt_w<<<(CC * CC + 255) / 256, 256, 0, stream>>>(wq, wqh, CC * CC);
  cvt_w<<<(CC * CC + 255) / 256, 256, 0, stream>>>(wk, wkh, CC * CC);
  cvt_w<<<(CC * CC + 255) / 256, 256, 0, stream>>>(wv, wvh, CC * CC);
  cvt_w<<<(CC * CC + 255) / 256, 256, 0, stream>>>(wp, wph, CC * CC);
  cvt_w<<<(HIDD * CC + 255) / 256, 256, 0, stream>>>(w1, w1h, HIDD * CC);
  cvt_w<<<(CC * HIDD + 255) / 256, 256, 0, stream>>>(w2, w2h, CC * HIDD);
  bn_prep<<<(CC + 255) / 256, 256, 0, stream>>>(qg, qb, qm, qv, nullptr, qsc, qof, CC);
  bn_prep<<<(CC + 255) / 256, 256, 0, stream>>>(kg, kb, km, kvv, nullptr, ksc, kof, CC);
  bn_prep<<<(CC + 255) / 256, 256, 0, stream>>>(vg, vb, vm, vvv, nullptr, vsc, vof, CC);
  bn_prep<<<(CC + 255) / 256, 256, 0, stream>>>(pg, pb, pm, pv, bp, psc, pof, CC);
  bn_prep<<<(HIDD + 255) / 256, 256, 0, stream>>>(g1, be1, m1, v1, b1, s1, o1, HIDD);
  bn_prep<<<(CC + 255) / 256, 256, 0, stream>>>(g2, be2, m2, v2, b2, s2, o2, CC);

  const dim3 gridC(NPAD / 16, CC / 128, BB);     // 24 o-tiles / 8 waves
  const dim3 gridH(NPAD / 16, HIDD / 128, BB);
  // q, k spikes (token-major); v spikes (channel-major)
  gemm_bn_lif<true,  false><<<gridC, 256, 0, stream>>>(wqh, xh, qsc, qof, nullptr, qsb, CC, CC, 1.0f);
  gemm_bn_lif<true,  false><<<gridC, 256, 0, stream>>>(wkh, xh, ksc, kof, nullptr, ksb, CC, CC, 1.0f);
  gemm_bn_lif<false, false><<<gridC, 256, 0, stream>>>(wvh, xh, vsc, vof, nullptr, vsb, CC, CC, 1.0f);
  // attention (mask + scale + LIF 0.5)
  attn_kernel<<<dim3(NPAD / 16, HHD / 4, BB), 128, 0, stream>>>(qsb, ksb, vsb, policy, oas);
  // proj + LIF + residual (x1 = x + o)
  gemm_bn_lif<true, true><<<gridC, 256, 0, stream>>>(wph, oas, psc, pof, xh, x1h, CC, CC, 1.0f);
  // MLP
  gemm_bn_lif<true, false><<<gridH, 256, 0, stream>>>(w1h, x1h, s1, o1, nullptr, h1s, CC, HIDD, 1.0f);
  gemm_bn_lif<true, false><<<gridC, 256, 0, stream>>>(w2h, h1s, s2, o2, nullptr, h2s, HIDD, CC, 1.0f);
  // out = x1 + h (transpose back to (T,B,C,N) f32)
  const int tot = TT * BB * CC * NN;
  final_add<<<(tot + 255) / 256, 256, 0, stream>>>(x1h, h2s, (float*)d_out, tot);
}
